// MultiHeadAttnLayer_28948079575832
// MI455X (gfx1250) — matrix operand order repaired and checked by reading, not compiled, not run on hardware
//
#include <hip/hip_runtime.h>
#include <hip/hip_bf16.h>

typedef __attribute__((ext_vector_type(16))) _Float16 v16h;
typedef __attribute__((ext_vector_type(8)))  float    v8f;

#define DD 64
#define HH 2
#define HD 128

__device__ __forceinline__ unsigned enc_f32(float f) {
  unsigned u = __float_as_uint(f);
  return (u & 0x80000000u) ? ~u : (u | 0x80000000u);
}
__device__ __forceinline__ float dec_f32(unsigned u) {
  unsigned v = (u & 0x80000000u) ? (u & 0x7FFFFFFFu) : ~u;
  return __uint_as_float(v);
}

// ---------------------------------------------------------------- init
__global__ void init_kernel(float* wv, float* denom, unsigned* menc,
                            int nwv, int nh) {
  int i = blockIdx.x * blockDim.x + threadIdx.x;
  if (i < nwv) wv[i] = 0.f;
  if (i < nh) { denom[i] = 0.f; menc[i] = 0u; }  // 0 < enc(any real) -> acts as -inf
}

// ------------------------------------------------- K/Q/V node GEMM (WMMA)
// out[n, h*64+f] = sum_d X[n,d] * W[h,f,d] + b[h,f]
// B matrix (64 x 128): B[d][col] = Wflat[col*64 + d]  (W flat IS col-major-T)
// Handles ONLY full 16-node tiles: no bounds checks anywhere.
__global__ __launch_bounds__(128) void kqv_kernel(
    const float* __restrict__ X,
    const float* __restrict__ Wk, const float* __restrict__ bk,
    const float* __restrict__ Wq, const float* __restrict__ bq,
    const float* __restrict__ Wv, const float* __restrict__ bv,
    float* __restrict__ Ko, float* __restrict__ Qo, float* __restrict__ Vo,
    int tilesFull) {
  __shared__ __align__(32) _Float16 wlds[3][HD][DD];  // [mat][col][d]  48 KB
  __shared__ float blds[3][HD];

  const float* Ws[3] = {Wk, Wq, Wv};
  const float* bs[3] = {bk, bq, bv};
  for (int m = 0; m < 3; ++m) {
    for (int i = threadIdx.x; i < HD * DD; i += 128)
      ((_Float16*)wlds[m])[i] = (_Float16)Ws[m][i];
    if (threadIdx.x < HD) blds[m][threadIdx.x] = bs[m][threadIdx.x];
  }
  __syncthreads();

  int wave = threadIdx.x >> 5;
  int lane = threadIdx.x & 31;
  int tile = blockIdx.x * 4 + wave;
  if (tile >= tilesFull) return;
  int node0 = tile * 16;

  // ---- A fragments: 16x32 f16, lane L holds row M=L&15.
  // lanes 0-15: halves 0..15 -> K=0..15
  // lanes 16-31: +16 on K base
  int mrow = lane & 15;
  int koff = (lane >> 4) * 16;
  const float* xr = X + (size_t)(node0 + mrow) * DD;
  v16h a0, a1;
#pragma unroll
  for (int i = 0; i < 16; ++i) {
    int k = i + koff;
    a0[i] = (_Float16)xr[k];
    a1[i] = (_Float16)xr[k + 32];
  }

  // ---- B fragments: 32x16 f16; lanes 0-15 -> K=k0..k0+15, lanes 16-31 -> +16
  int ncol = lane & 15;
  int kb = (lane >> 4) * 16;
  int mbase = (lane >> 4) * 8;

  // Unrolled over the 3 output matrices so each store keeps its direct
  // global pointer (global_store_b32, not flat_store: flat also ticks DScnt
  // and would serialize against the ds_load_b128 B-fragment stream).
#pragma unroll
  for (int m = 0; m < 3; ++m) {
    float* __restrict__ Om = (m == 0) ? Ko : (m == 1) ? Qo : Vo;
    for (int ct = 0; ct < 8; ++ct) {
      int col = ct * 16 + ncol;
      const v16h b0 = *(const v16h*)&wlds[m][col][kb];
      const v16h b1 = *(const v16h*)&wlds[m][col][32 + kb];
      v8f c = {};
      c = __builtin_amdgcn_wmma_f32_16x16x32_f16(false, a0, false, b0,
                                                 (short)0, c, false, false);
      c = __builtin_amdgcn_wmma_f32_16x16x32_f16(false, a1, false, b1,
                                                 (short)0, c, false, false);
      float bias = blds[m][col];
      float* outp = Om + ((size_t)node0 + mbase) * HD + col;
#pragma unroll
      for (int i = 0; i < 8; ++i)
        outp[(size_t)i * HD] = c[i] + bias;
    }
  }
}

// ------------------------------------- scalar tail for K/Q/V (rem nodes)
__global__ void kqv_tail_kernel(const float* __restrict__ X,
                                const float* __restrict__ Wk,
                                const float* __restrict__ bk,
                                const float* __restrict__ Wq,
                                const float* __restrict__ bq,
                                const float* __restrict__ Wv,
                                const float* __restrict__ bv,
                                float* __restrict__ Ko, float* __restrict__ Qo,
                                float* __restrict__ Vo, int node0, int rem) {
  int t = blockIdx.x * blockDim.x + threadIdx.x;  // (mat, node, col)
  if (t >= rem * HD * 3) return;
  int m = t / (rem * HD);
  int r = t - m * rem * HD;
  int n = node0 + r / HD;
  int col = r % HD;
  const float* W = (m == 0) ? Wk : (m == 1) ? Wq : Wv;
  const float* b = (m == 0) ? bk : (m == 1) ? bq : bv;
  float* O = (m == 0) ? Ko : (m == 1) ? Qo : Vo;
  const float* x = X + (size_t)n * DD;
  const float* w = W + (size_t)col * DD;
  float acc = b[col];
  for (int d = 0; d < DD; ++d) acc += x[d] * w[d];
  O[(size_t)n * HD + col] = acc;
}

// ----------------------------------------------- per-edge score + seg-max
__global__ void score_kernel(const float* __restrict__ K,
                             const float* __restrict__ Q,
                             const int* __restrict__ src,
                             const int* __restrict__ dst,
                             float* __restrict__ score,
                             unsigned* __restrict__ menc, int E) {
  int e = blockIdx.x * blockDim.x + threadIdx.x;
  if (e >= E) return;
  int s = src[e], d = dst[e];
  const float4* kr = (const float4*)(K + (size_t)s * HD);
  const float4* qr = (const float4*)(Q + (size_t)d * HD);
#pragma unroll
  for (int h = 0; h < HH; ++h) {
    float acc = 0.f;
#pragma unroll
    for (int i = 0; i < 16; ++i) {
      float4 kv = kr[h * 16 + i], qv = qr[h * 16 + i];
      acc += kv.x * qv.x + kv.y * qv.y + kv.z * qv.z + kv.w * qv.w;
    }
    score[(size_t)e * HH + h] = acc;
    atomicMax(&menc[(size_t)d * HH + h], enc_f32(acc));
  }
}

// -------------------------------------------------------- decode seg-max
__global__ void decode_kernel(const unsigned* __restrict__ menc,
                              float* __restrict__ m, int nh) {
  int i = blockIdx.x * blockDim.x + threadIdx.x;
  if (i >= nh) return;
  float f = dec_f32(menc[i]);
  m[i] = isfinite(f) ? f : 0.f;  // nodes with no in-edges -> 0 (as reference)
}

// ------------------------------------------------------ exp + denominator
__global__ void exp_kernel(float* __restrict__ score,
                           const float* __restrict__ m,
                           float* __restrict__ denom,
                           const int* __restrict__ dst, int E) {
  int e = blockIdx.x * blockDim.x + threadIdx.x;
  if (e >= E) return;
  int d = dst[e];
#pragma unroll
  for (int h = 0; h < HH; ++h) {
    float es = expf(score[(size_t)e * HH + h] - m[(size_t)d * HH + h]);
    score[(size_t)e * HH + h] = es;
    atomicAdd(&denom[(size_t)d * HH + h], es);
  }
}

// -------------------------------------- warp-per-edge weighted-V scatter
__global__ __launch_bounds__(128) void scatter_kernel(
    const float* __restrict__ V, const float* __restrict__ es,
    const float* __restrict__ denom, const int* __restrict__ src,
    const int* __restrict__ dst, float* __restrict__ wv,
    float* __restrict__ edge_w, int E) {
  int wid = (int)(((size_t)blockIdx.x * blockDim.x + threadIdx.x) >> 5);
  int lane = threadIdx.x & 31;
  if (wid >= E) return;
  int s = src[wid], d = dst[wid];
  float w0 = es[(size_t)wid * HH + 0] / denom[(size_t)d * HH + 0];
  float w1 = es[(size_t)wid * HH + 1] / denom[(size_t)d * HH + 1];
  if (lane == 0) edge_w[wid] = 0.5f * (w0 + w1);
  float4 vv = ((const float4*)(V + (size_t)s * HD))[lane];
  float w = (lane < 16) ? w0 : w1;  // lanes 0-15 cover head 0 cols, 16-31 head 1
  float* out = wv + (size_t)d * HD + lane * 4;
  atomicAdd(out + 0, w * vv.x);
  atomicAdd(out + 1, w * vv.y);
  atomicAdd(out + 2, w * vv.z);
  atomicAdd(out + 3, w * vv.w);
}

// --------------------------------------------------- O projection (WMMA)
// out[n,o] = sum_f wv[n,f] * Wo[o,f] + bo[o];  B[f][o] = Woflat[o*128+f]
// Handles ONLY full 16-node tiles.
__global__ __launch_bounds__(128) void out_kernel(
    const float* __restrict__ WV, const float* __restrict__ Wo,
    const float* __restrict__ bo, float* __restrict__ OUT, int tilesFull) {
  __shared__ __align__(32) _Float16 wlds[DD][HD];  // [o][f]  16 KB
  __shared__ float blds[DD];
  for (int i = threadIdx.x; i < DD * HD; i += 128)
    ((_Float16*)wlds)[i] = (_Float16)Wo[i];
  if (threadIdx.x < DD) blds[threadIdx.x] = bo[threadIdx.x];
  __syncthreads();

  int wave = threadIdx.x >> 5;
  int lane = threadIdx.x & 31;
  int tile = blockIdx.x * 4 + wave;
  if (tile >= tilesFull) return;
  int node0 = tile * 16;

  int mrow = lane & 15;
  int koff = (lane >> 4) * 16;
  const float* xr = WV + (size_t)(node0 + mrow) * HD;
  v16h a[4];
#pragma unroll
  for (int f = 0; f < 4; ++f)
#pragma unroll
    for (int i = 0; i < 16; ++i) {
      int k = f * 32 + i + koff;
      a[f][i] = (_Float16)xr[k];
    }

  int ncol = lane & 15;
  int kb = (lane >> 4) * 16;
  int mbase = (lane >> 4) * 8;
  for (int ct = 0; ct < 4; ++ct) {
    int col = ct * 16 + ncol;
    v8f c = {};
#pragma unroll
    for (int f = 0; f < 4; ++f) {
      const v16h b = *(const v16h*)&wlds[col][f * 32 + kb];
      c = __builtin_amdgcn_wmma_f32_16x16x32_f16(false, a[f], false, b,
                                                 (short)0, c, false, false);
    }
    float bias = blds[col];
    float* outp = OUT + ((size_t)node0 + mbase) * DD + col;
#pragma unroll
    for (int i = 0; i < 8; ++i)
      outp[(size_t)i * DD] = c[i] + bias;
  }
}

// ------------------------------------------ scalar tail for O projection
__global__ void out_tail_kernel(const float* __restrict__ WV,
                                const float* __restrict__ Wo,
                                const float* __restrict__ bo,
                                float* __restrict__ OUT, int node0, int rem) {
  int t = blockIdx.x * blockDim.x + threadIdx.x;  // (node, o)
  if (t >= rem * DD) return;
  int n = node0 + t / DD;
  int o = t % DD;
  const float* x = WV + (size_t)n * HD;
  const float* w = Wo + (size_t)o * HD;
  float acc = bo[o];
  for (int f = 0; f < HD; ++f) acc += x[f] * w[f];
  OUT[(size_t)n * DD + o] = acc;
}

extern "C" void kernel_launch(void* const* d_in, const int* in_sizes, int n_in,
                              void* d_out, int out_size, void* d_ws,
                              size_t ws_size, hipStream_t stream) {
  const float* X  = (const float*)d_in[0];
  const float* Wk = (const float*)d_in[1];
  const float* bk = (const float*)d_in[2];
  const float* Wq = (const float*)d_in[3];
  const float* bq = (const float*)d_in[4];
  const float* Wv = (const float*)d_in[5];
  const float* bv = (const float*)d_in[6];
  const float* Wo = (const float*)d_in[7];
  const float* bo = (const float*)d_in[8];
  const int* src  = (const int*)d_in[9];
  const int* dst  = (const int*)d_in[10];
  int N = in_sizes[0] / DD;
  int E = in_sizes[9];

  // workspace layout (floats): K, Q, V, WV [N*128 each], score [E*2],
  // denom [N*2], mdec [N*2], menc [N*2 uint]
  float* Kb    = (float*)d_ws;
  float* Qb    = Kb + (size_t)N * HD;
  float* Vb    = Qb + (size_t)N * HD;
  float* WVb   = Vb + (size_t)N * HD;
  float* score = WVb + (size_t)N * HD;
  float* denom = score + (size_t)E * HH;
  float* mdec  = denom + (size_t)N * HH;
  unsigned* menc = (unsigned*)(mdec + (size_t)N * HH);

  float* out_nf = (float*)d_out;
  float* edge_w = out_nf + (size_t)N * DD;

  int nwv = N * HD, nh = N * HH;
  init_kernel<<<(nwv + 255) / 256, 256, 0, stream>>>(WVb, denom, menc, nwv, nh);

  int tilesFull = N / 16;           // full 16-node tiles (WMMA path)
  int rem = N - tilesFull * 16;     // leftover nodes (scalar path; 0 here)
  int gb = (tilesFull + 3) / 4;

  if (tilesFull)
    kqv_kernel<<<gb, 128, 0, stream>>>(X, Wk, bk, Wq, bq, Wv, bv, Kb, Qb, Vb,
                                       tilesFull);
  if (rem) {
    int tt = rem * HD * 3;
    kqv_tail_kernel<<<(tt + 127) / 128, 128, 0, stream>>>(
        X, Wk, bk, Wq, bq, Wv, bv, Kb, Qb, Vb, tilesFull * 16, rem);
  }

  score_kernel<<<(E + 255) / 256, 256, 0, stream>>>(Kb, Qb, src, dst, score,
                                                    menc, E);
  decode_kernel<<<(nh + 255) / 256, 256, 0, stream>>>(menc, mdec, nh);
  exp_kernel<<<(E + 255) / 256, 256, 0, stream>>>(score, mdec, denom, dst, E);

  int sblocks = (int)(((size_t)E * 32 + 127) / 128);
  scatter_kernel<<<sblocks, 128, 0, stream>>>(Vb, score, denom, src, dst, WVb,
                                              edge_w, E);

  if (tilesFull)
    out_kernel<<<gb, 128, 0, stream>>>(WVb, Wo, bo, out_nf, tilesFull);
  if (rem) {
    int tt = rem * DD;
    out_tail_kernel<<<(tt + 127) / 128, 128, 0, stream>>>(
        WVb, Wo, bo, out_nf, tilesFull * 16, rem);
  }
}
